// ExtractorMLP_84507776516382
// MI455X (gfx1250) — matrix operand order repaired and checked by reading, not compiled, not run on hardware
//
#include <hip/hip_runtime.h>

// ---------------------------------------------------------------------------
// MI455X (gfx1250) ExtractorMLP: edge-gather -> [128x256 GEMM + inst.norm +
// relu] -> [256x64 GEMM + inst.norm + relu] -> [64x1 dot], E=1.6M, 128 graphs.
// bf16 V_WMMA_F32_16X16X32_BF16 GEMMs with f32 accumulation; edges counting-
// sorted by segment so per-(seg,channel) statistics accumulate in registers.
// ---------------------------------------------------------------------------

typedef __bf16 bf16_t;
typedef __attribute__((ext_vector_type(16))) __bf16 v16bf;
typedef __attribute__((ext_vector_type(8)))  __bf16 v8bf;
typedef __attribute__((ext_vector_type(8)))  float  v8f;

#define SHUF16(lo, hi) __builtin_shufflevector(lo, hi, 0,1,2,3,4,5,6,7,8,9,10,11,12,13,14,15)

__device__ __forceinline__ unsigned short f2bf(float f) {
  unsigned int u = __builtin_bit_cast(unsigned int, f);
  unsigned int r = u + 0x7FFFu + ((u >> 16) & 1u);   // round-to-nearest-even
  return (unsigned short)(r >> 16);
}
__device__ __forceinline__ float bf2f(unsigned short h) {
  unsigned int u = ((unsigned int)h) << 16;
  return __builtin_bit_cast(float, u);
}

// ---------------------------------------------------------------------------
// init: perm = -1 over padded capacity; zero stats + seg counts
// ---------------------------------------------------------------------------
__global__ void k_init(int* perm, long long cap,
                       float* s1S, float* s1Q, float* s2S, float* s2Q,
                       int* segCount) {
  long long i = (long long)blockIdx.x * blockDim.x + threadIdx.x;
  long long stride = (long long)gridDim.x * blockDim.x;
  for (long long x = i; x < cap; x += stride) perm[x] = -1;
  for (long long x = i; x < 32768; x += stride) { s1S[x] = 0.f; s1Q[x] = 0.f; }
  for (long long x = i; x < 8192;  x += stride) { s2S[x] = 0.f; s2Q[x] = 0.f; }
  for (long long x = i; x < 128;   x += stride) segCount[x] = 0;
}

// ---------------------------------------------------------------------------
// histogram of seg = batch[col[e]]
// ---------------------------------------------------------------------------
__global__ void k_hist(const int* __restrict__ eidx, const int* __restrict__ batch,
                       int* __restrict__ segCount, int E) {
  __shared__ int h[128];
  int tid = threadIdx.x;
  if (tid < 128) h[tid] = 0;
  __syncthreads();
  for (long long i = (long long)blockIdx.x * blockDim.x + tid; i < E;
       i += (long long)gridDim.x * blockDim.x) {
    int s = batch[eidx[i]] & 127;
    atomicAdd(&h[s], 1);
  }
  __syncthreads();
  if (tid < 128 && h[tid]) atomicAdd(&segCount[tid], h[tid]);
}

// ---------------------------------------------------------------------------
// scan: per-seg padded starts (16-row tile aligned), cursors, total tile count
// ---------------------------------------------------------------------------
__global__ void k_scan(const int* __restrict__ segCount, int* segStart,
                       int* segCursor, int* totalTiles) {
  if (threadIdx.x == 0 && blockIdx.x == 0) {
    int acc = 0;
    for (int s = 0; s < 128; ++s) {
      segStart[s] = acc;
      segCursor[s] = acc;
      int tiles = (segCount[s] + 15) >> 4;
      acc += tiles * 16;
    }
    segStart[128] = acc;
    *totalTiles = acc >> 4;
  }
}

// ---------------------------------------------------------------------------
// scatter edges into seg-sorted padded layout (block-aggregated atomics)
// ---------------------------------------------------------------------------
__global__ void k_scatter(const int* __restrict__ eidx, const int* __restrict__ batch,
                          int* __restrict__ segCursor, int* __restrict__ perm, int E) {
  __shared__ int cnt[128];
  __shared__ int base[128];
  int tid = threadIdx.x;
  if (tid < 128) cnt[tid] = 0;
  __syncthreads();
  long long i = (long long)blockIdx.x * blockDim.x + tid;
  int s = 0, r = 0;
  bool ok = (i < E);
  if (ok) { s = batch[eidx[i]] & 127; r = atomicAdd(&cnt[s], 1); }
  __syncthreads();
  if (tid < 128 && cnt[tid] > 0) base[tid] = atomicAdd(&segCursor[tid], cnt[tid]);
  __syncthreads();
  if (ok) perm[base[s] + r] = (int)i;
}

// ---------------------------------------------------------------------------
// seg id per 16-row tile (binary search over padded starts)
// ---------------------------------------------------------------------------
__global__ void k_segtile(const int* __restrict__ segStart, int* __restrict__ segOfTile,
                          const int* __restrict__ totalTilesPtr, int maxTiles) {
  int t = blockIdx.x * 256 + threadIdx.x;
  if (t >= maxTiles) return;
  int total = *totalTilesPtr;
  int seg = 0;
  if (t < total) {
    int pos = t * 16, lo = 0, hi = 127;
    while (lo < hi) {
      int mid = (lo + hi + 1) >> 1;
      if (segStart[mid] <= pos) lo = mid; else hi = mid - 1;
    }
    seg = lo;
  }
  segOfTile[t] = seg;
}

// ---------------------------------------------------------------------------
// convert W1 (128x256) and W2 (256x64) to transposed bf16: wT[n][k]
// ---------------------------------------------------------------------------
__global__ void k_cvtw(const float* __restrict__ W1, const float* __restrict__ W2,
                       unsigned short* __restrict__ w1t, unsigned short* __restrict__ w2t) {
  int i = blockIdx.x * 256 + threadIdx.x;
  if (i < 256 * 128) {
    int n = i >> 7, k = i & 127;
    w1t[i] = f2bf(W1[k * 256 + n]);
  }
  int j = i - 256 * 128;
  if (j >= 0 && j < 64 * 256) {
    int n = j >> 8, k = j & 255;
    w2t[j] = f2bf(W2[k * 64 + n]);
  }
}

// ---------------------------------------------------------------------------
// GEMM1: A = bf16(concat(emb[col],emb[row])) [16x128], B = W1t (LDS, 64KB)
// C tile 16x256 per block iteration, wave w -> N-tiles 2w, 2w+1.
// Register per-channel sum/sumsq accumulation (single-seg tiles), Y1 bf16 out.
// ---------------------------------------------------------------------------
__global__ void k_gemm1(const float* __restrict__ emb, const int* __restrict__ eidx,
                        const float* __restrict__ b1g,
                        const int* __restrict__ perm, const int* __restrict__ segOfTile,
                        const int* __restrict__ totalTilesPtr,
                        const unsigned short* __restrict__ w1t,
                        unsigned short* __restrict__ Y1,
                        float* __restrict__ s1S, float* __restrict__ s1Q,
                        int E, int tilesPerBlk) {
  __shared__ alignas(32) unsigned short sW[256 * 128];  // W1^T bf16, 64KB
  __shared__ alignas(32) unsigned short sA[16 * 128];   // A tile bf16, 4KB
  __shared__ alignas(32) unsigned short sC[16 * 256];   // C staging bf16, 8KB
  __shared__ float sB1[256];
  __shared__ int sValid[16];

  int tid = threadIdx.x;
  {  // stage W1^T + bias
    const uint4* src = (const uint4*)w1t;
    uint4* dst = (uint4*)sW;
    for (int i = tid; i < 256 * 128 * 2 / 16; i += 256) dst[i] = src[i];
    for (int i = tid; i < 256; i += 256) sB1[i] = b1g[i];
  }
  __syncthreads();

  int total = *totalTilesPtr;
  int t0 = blockIdx.x * tilesPerBlk;
  int t1 = t0 + tilesPerBlk; if (t1 > total) t1 = total;

  int wave = tid >> 5, lane = tid & 31, half = lane >> 4, lr = lane & 15;
  int n0 = wave * 2, n1 = wave * 2 + 1;

  int curSeg = -1;
  float aS0 = 0.f, aQ0 = 0.f, aS1 = 0.f, aQ1 = 0.f;

  for (int t = t0; t < t1; ++t) {
    int seg = segOfTile[t];
    if (seg != curSeg) {
      if (curSeg >= 0 && half == 0) {
        atomicAdd(&s1S[curSeg * 256 + n0 * 16 + lr], aS0);
        atomicAdd(&s1Q[curSeg * 256 + n0 * 16 + lr], aQ0);
        atomicAdd(&s1S[curSeg * 256 + n1 * 16 + lr], aS1);
        atomicAdd(&s1Q[curSeg * 256 + n1 * 16 + lr], aQ1);
      }
      aS0 = aQ0 = aS1 = aQ1 = 0.f;
      curSeg = seg;
    }
    long long posBase = (long long)t * 16;

    {  // gather A: row m = tid/16, 16 threads per row, 8 floats each
      int m = tid >> 4, p = tid & 15;
      int e = perm[posBase + m];
      if (p == 0) sValid[m] = (e >= 0) ? 1 : 0;
      float4 v0 = make_float4(0.f, 0.f, 0.f, 0.f), v1 = v0;
      if (e >= 0) {
        int node = (p < 8) ? eidx[e] : eidx[E + e];
        const float* src = emb + (long long)node * 64 + (p & 7) * 8;
        v0 = ((const float4*)src)[0];
        v1 = ((const float4*)src)[1];
      }
      unsigned short* d = &sA[m * 128 + p * 8];
      d[0] = f2bf(v0.x); d[1] = f2bf(v0.y); d[2] = f2bf(v0.z); d[3] = f2bf(v0.w);
      d[4] = f2bf(v1.x); d[5] = f2bf(v1.y); d[6] = f2bf(v1.z); d[7] = f2bf(v1.w);
    }
    __syncthreads();

    v8f c0 = {}; v8f c1 = {};
#pragma unroll
    for (int ks = 0; ks < 4; ++ks) {
      int kb = ks * 32;
      // A frag: lanes<16 hold K kb..kb+15, lanes>=16 hold kb+16..kb+31
      v16bf a = *(const v16bf*)&sA[lr * 128 + kb + half * 16];
      // B frag: lanes<16 hold K kb..kb+15, lanes>=16 hold kb+16..kb+31
      v16bf bb0 = *(const v16bf*)&sW[(n0 * 16 + lr) * 128 + kb + half * 16];
      v16bf bb1 = *(const v16bf*)&sW[(n1 * 16 + lr) * 128 + kb + half * 16];
      c0 = __builtin_amdgcn_wmma_f32_16x16x32_bf16(false, a, false, bb0, (short)0, c0, false, false);
      c1 = __builtin_amdgcn_wmma_f32_16x16x32_bf16(false, a, false, bb1, (short)0, c1, false, false);
    }

    float bias0 = sB1[n0 * 16 + lr], bias1 = sB1[n1 * 16 + lr];
    float s0 = 0.f, q0 = 0.f, s1v = 0.f, q1v = 0.f;
#pragma unroll
    for (int j = 0; j < 8; ++j) {
      int m = half * 8 + j;
      float y0 = c0[j] + bias0;
      float y1 = c1[j] + bias1;
      if (sValid[m]) { s0 += y0; q0 += y0 * y0; s1v += y1; q1v += y1 * y1; }
      sC[m * 256 + n0 * 16 + lr] = f2bf(y0);
      sC[m * 256 + n1 * 16 + lr] = f2bf(y1);
    }
    s0 += __shfl_xor(s0, 16, 32);  q0 += __shfl_xor(q0, 16, 32);
    s1v += __shfl_xor(s1v, 16, 32); q1v += __shfl_xor(q1v, 16, 32);
    aS0 += s0; aQ0 += q0; aS1 += s1v; aQ1 += q1v;
    __syncthreads();

    {  // coalesced Y1 store: 16 rows x 512B, 32B per thread
      int m = tid >> 4, ch = (tid & 15) * 16;
      const uint4* s = (const uint4*)&sC[m * 256 + ch];
      uint4 x0 = s[0], x1 = s[1];
      uint4* d = (uint4*)&Y1[(posBase + m) * 256 + ch];
      d[0] = x0; d[1] = x1;
    }
  }
  if (curSeg >= 0 && half == 0) {
    atomicAdd(&s1S[curSeg * 256 + n0 * 16 + lr], aS0);
    atomicAdd(&s1Q[curSeg * 256 + n0 * 16 + lr], aQ0);
    atomicAdd(&s1S[curSeg * 256 + n1 * 16 + lr], aS1);
    atomicAdd(&s1Q[curSeg * 256 + n1 * 16 + lr], aQ1);
  }
}

// ---------------------------------------------------------------------------
// stats -> (scale a, shift b): (y-mean)*rsqrt(var+eps) = a*y + b
// ---------------------------------------------------------------------------
__global__ void k_norm(const float* __restrict__ S, const float* __restrict__ Q,
                       const int* __restrict__ segCount,
                       float* __restrict__ A, float* __restrict__ B, int C) {
  int i = blockIdx.x * 256 + threadIdx.x;
  if (i >= 128 * C) return;
  int seg = i / C;
  int cnt = segCount[seg];
  float n = (float)(cnt > 1 ? cnt : 1);
  float mean = S[i] / n;
  float var = Q[i] / n - mean * mean;
  var = var > 0.f ? var : 0.f;
  float a = rsqrtf(var + 1e-5f);
  A[i] = a;
  B[i] = -mean * a;
}

// ---------------------------------------------------------------------------
// GEMM2: A = bf16(relu(a1*Y1+b1)) [32x256 per iter, 2 M-tiles], B = W2t (LDS)
// wave w -> (Mtile w/4, Ntile w%4); Y2 bf16 out + layer-2 stats.
// ---------------------------------------------------------------------------
__global__ void k_gemm2(const unsigned short* __restrict__ Y1,
                        const unsigned short* __restrict__ w2t,
                        const float* __restrict__ b2g,
                        const float* __restrict__ n1A, const float* __restrict__ n1B,
                        const int* __restrict__ perm, const int* __restrict__ segOfTile,
                        const int* __restrict__ totalTilesPtr,
                        unsigned short* __restrict__ Y2,
                        float* __restrict__ s2S, float* __restrict__ s2Q,
                        int pairsPerBlk, long long cap) {
  __shared__ alignas(32) unsigned short sW[64 * 256];  // W2^T bf16, 32KB
  __shared__ alignas(32) unsigned short sA[32 * 256];  // 16KB
  __shared__ alignas(32) unsigned short sC[32 * 64];   // 4KB
  __shared__ int sValid[32];

  int tid = threadIdx.x;
  {
    const uint4* src = (const uint4*)w2t;
    uint4* dst = (uint4*)sW;
    for (int i = tid; i < 64 * 256 * 2 / 16; i += 256) dst[i] = src[i];
  }
  __syncthreads();

  int total = *totalTilesPtr;
  int wave = tid >> 5, lane = tid & 31, half = lane >> 4, lr = lane & 15;
  int msub = wave >> 2, nt = wave & 3;
  float bias = b2g[nt * 16 + lr];
  int curSeg = -1;
  float aS = 0.f, aQ = 0.f;

  long long pr0 = (long long)blockIdx.x * pairsPerBlk;
  for (long long pr = pr0; pr < pr0 + pairsPerBlk; ++pr) {
    long long t = pr * 2;
    if (t >= total) break;
    long long posBase = t * 16;
    int segSub = segOfTile[(t + msub < total) ? (t + msub) : t];

    {  // A load with norm1 + relu: row m = tid/8, 32 channels per thread
      int m = tid >> 3, p = tid & 7;
      long long pos = posBase + m;
      int e = (pos < cap) ? perm[pos] : -1;
      if (p == 0) sValid[m] = (e >= 0) ? 1 : 0;
      int rowTile = (m < 16) ? 0 : 1;
      int segR = segOfTile[(t + rowTile < total) ? (t + rowTile) : t];
      unsigned short* d = &sA[m * 256 + p * 32];
      if (e >= 0) {
        const unsigned short* src = &Y1[pos * 256 + p * 32];
        const float* na = &n1A[segR * 256 + p * 32];
        const float* nb = &n1B[segR * 256 + p * 32];
#pragma unroll
        for (int i = 0; i < 32; ++i) {
          float v = bf2f(src[i]);
          float y = na[i] * v + nb[i];
          d[i] = f2bf(y > 0.f ? y : 0.f);
        }
      } else {
#pragma unroll
        for (int i = 0; i < 32; ++i) d[i] = 0;
      }
    }
    __syncthreads();

    if (segSub != curSeg) {
      if (curSeg >= 0 && half == 0) {
        atomicAdd(&s2S[curSeg * 64 + nt * 16 + lr], aS);
        atomicAdd(&s2Q[curSeg * 64 + nt * 16 + lr], aQ);
      }
      aS = 0.f; aQ = 0.f;
      curSeg = segSub;
    }

    v8f c = {};
#pragma unroll
    for (int ks = 0; ks < 8; ++ks) {
      int kb = ks * 32;
      v16bf a = *(const v16bf*)&sA[(msub * 16 + lr) * 256 + kb + half * 16];
      v16bf b = *(const v16bf*)&sW[(nt * 16 + lr) * 256 + kb + half * 16];
      c = __builtin_amdgcn_wmma_f32_16x16x32_bf16(false, a, false, b, (short)0, c, false, false);
    }

    float s = 0.f, q = 0.f;
#pragma unroll
    for (int j = 0; j < 8; ++j) {
      int m = msub * 16 + half * 8 + j;
      float y = c[j] + bias;
      if (sValid[m]) { s += y; q += y * y; }
      sC[m * 64 + nt * 16 + lr] = f2bf(y);
    }
    s += __shfl_xor(s, 16, 32);
    q += __shfl_xor(q, 16, 32);
    aS += s; aQ += q;
    __syncthreads();

    {  // Y2 store: 32 rows x 128B, 16B per thread
      int m = tid >> 3, ck = (tid & 7) * 8;
      long long pos = posBase + m;
      if (pos < cap) {
        uint4 x = *(const uint4*)&sC[m * 64 + ck];
        *(uint4*)&Y2[pos * 64 + ck] = x;
      }
    }
  }
  if (curSeg >= 0 && half == 0) {
    atomicAdd(&s2S[curSeg * 64 + nt * 16 + lr], aS);
    atomicAdd(&s2Q[curSeg * 64 + nt * 16 + lr], aQ);
  }
}

// ---------------------------------------------------------------------------
// final layer: out[e] = relu(a2*Y2+b2) . W3 + b3, scattered to original order
// ---------------------------------------------------------------------------
__global__ void k_out(const unsigned short* __restrict__ Y2,
                      const int* __restrict__ perm, const int* __restrict__ segOfTile,
                      const float* __restrict__ n2A, const float* __restrict__ n2B,
                      const float* __restrict__ W3, const float* __restrict__ b3,
                      float* __restrict__ out, long long cap) {
  __shared__ float sW3[64];
  int tid = threadIdx.x;
  if (tid < 64) sW3[tid] = W3[tid];
  __syncthreads();
  long long p = (long long)blockIdx.x * blockDim.x + tid;
  if (p >= cap) return;
  int e = perm[p];
  if (e < 0) return;
  int seg = segOfTile[p >> 4];
  const unsigned short* src = &Y2[p * 64];
  const float* na = &n2A[seg * 64];
  const float* nb = &n2B[seg * 64];
  float acc = 0.f;
#pragma unroll
  for (int c = 0; c < 64; ++c) {
    float v = bf2f(src[c]);
    float y = na[c] * v + nb[c];
    y = y > 0.f ? y : 0.f;
    acc += y * sW3[c];
  }
  out[e] = acc + b3[0];
}

// ---------------------------------------------------------------------------
// host launcher
// ---------------------------------------------------------------------------
extern "C" void kernel_launch(void* const* d_in, const int* in_sizes, int n_in,
                              void* d_out, int out_size, void* d_ws, size_t ws_size,
                              hipStream_t stream) {
  const float* emb   = (const float*)d_in[0];
  const int*   eidx  = (const int*)d_in[1];
  const int*   batch = (const int*)d_in[2];
  const float* W1    = (const float*)d_in[3];
  const float* b1    = (const float*)d_in[4];
  const float* W2    = (const float*)d_in[5];
  const float* b2    = (const float*)d_in[6];
  const float* W3    = (const float*)d_in[7];
  const float* b3    = (const float*)d_in[8];
  float* out = (float*)d_out;

  const int E = in_sizes[1] / 2;
  const int maxTiles = (E + 15) / 16 + 128;      // per-seg padding slack
  const long long cap = (long long)maxTiles * 16;

  char* w = (char*)d_ws;
  auto alloc = [&](size_t bytes) -> char* {
    char* p = w;
    w += (bytes + 255) & ~(size_t)255;
    return p;
  };
  int*   perm       = (int*)alloc((size_t)cap * 4);
  int*   segCount   = (int*)alloc(128 * 4);
  int*   segStart   = (int*)alloc(129 * 4);
  int*   segCursor  = (int*)alloc(128 * 4);
  int*   totalTiles = (int*)alloc(4);
  int*   segOfTile  = (int*)alloc((size_t)maxTiles * 4);
  float* s1S = (float*)alloc(32768 * 4);
  float* s1Q = (float*)alloc(32768 * 4);
  float* n1A = (float*)alloc(32768 * 4);
  float* n1B = (float*)alloc(32768 * 4);
  float* s2S = (float*)alloc(8192 * 4);
  float* s2Q = (float*)alloc(8192 * 4);
  float* n2A = (float*)alloc(8192 * 4);
  float* n2B = (float*)alloc(8192 * 4);
  unsigned short* w1t = (unsigned short*)alloc(256 * 128 * 2);
  unsigned short* w2t = (unsigned short*)alloc(64 * 256 * 2);
  unsigned short* Y1  = (unsigned short*)alloc((size_t)cap * 256 * 2);
  unsigned short* Y2  = (unsigned short*)alloc((size_t)cap * 64 * 2);

  const int TB = 256;
  int gInit = (int)((cap + TB - 1) / TB);
  k_init<<<gInit, TB, 0, stream>>>(perm, cap, s1S, s1Q, s2S, s2Q, segCount);

  int gHist = (E + TB * 8 - 1) / (TB * 8);
  if (gHist < 1) gHist = 1;
  k_hist<<<gHist, TB, 0, stream>>>(eidx, batch, segCount, E);

  k_scan<<<1, 1, 0, stream>>>(segCount, segStart, segCursor, totalTiles);

  int gScat = (E + TB - 1) / TB;
  k_scatter<<<gScat, TB, 0, stream>>>(eidx, batch, segCursor, perm, E);

  int gSegT = (maxTiles + TB - 1) / TB;
  k_segtile<<<gSegT, TB, 0, stream>>>(segStart, segOfTile, totalTiles, maxTiles);

  int gCvt = (256 * 128 + 64 * 256 + TB - 1) / TB;
  k_cvtw<<<gCvt, TB, 0, stream>>>(W1, W2, w1t, w2t);

  int NB1 = 2048;
  if (NB1 > maxTiles) NB1 = maxTiles;
  int tilesPerBlk = (maxTiles + NB1 - 1) / NB1;
  k_gemm1<<<NB1, TB, 0, stream>>>(emb, eidx, b1, perm, segOfTile, totalTiles,
                                  w1t, Y1, s1S, s1Q, E, tilesPerBlk);

  k_norm<<<128, TB, 0, stream>>>(s1S, s1Q, segCount, n1A, n1B, 256);

  long long pairs = (maxTiles + 1) / 2;
  int NB2 = 2048;
  if ((long long)NB2 > pairs) NB2 = (int)pairs;
  int pairsPerBlk = (int)((pairs + NB2 - 1) / NB2);
  k_gemm2<<<NB2, TB, 0, stream>>>(Y1, w2t, b2, n1A, n1B, perm, segOfTile,
                                  totalTiles, Y2, s2S, s2Q, pairsPerBlk, cap);

  k_norm<<<32, TB, 0, stream>>>(s2S, s2Q, segCount, n2A, n2B, 64);

  int gOut = (int)((cap + TB - 1) / TB);
  k_out<<<gOut, TB, 0, stream>>>(Y2, perm, segOfTile, n2A, n2B, W3, b3, out, cap);
}
